// DragnnLSTMCell_20856361189885
// MI455X (gfx1250) — compile-verified, matrix operand order checked by reading, not run on hardware
//
#include <hip/hip_runtime.h>
#include <hip/hip_bf16.h>

// ---------------------------------------------------------------------------
// DragnnLSTMCell on gfx1250 (MI455X).
//   Stage 1: convert X and all weight matrices fp32 -> bf16 (round-nearest-even).
//   Stage 2: WMMA GEMM:  XW[g] = X_bf16 @ Wx_bf16[g] + bias[g]   (g = i,o,c)
//            v_wmma_f32_16x16x32_bf16, 64x64 block tile, double-buffered LDS,
//            A-tile staged with global_load_async_to_lds_b128 (ASYNCcnt).
//   Stage 3: persistent recurrent kernel, 1024 sequential steps, grid barriers.
// ---------------------------------------------------------------------------

typedef unsigned short u16;
typedef __attribute__((ext_vector_type(16))) __bf16 v16bf;
typedef __attribute__((ext_vector_type(8)))  float  v8f;

static constexpr int S = 1024;   // SEQ_LEN
static constexpr int H = 2048;   // HIDDEN_DIM == INPUT_DIM
static constexpr int NB = 64;    // persistent blocks for recurrent kernel

struct __attribute__((aligned(16))) U128 { unsigned x, y, z, w; };
union FragU { U128 u[2]; v16bf v; };

__device__ __forceinline__ float bf2f(u16 v) {
    return __uint_as_float(((unsigned)v) << 16);
}

// LDS byte offset of a __shared__ object: AS3->generic cast places the LDS
// offset in the low 32 bits of the flat address.
__device__ __forceinline__ unsigned lds_off(const void* p) {
    return (unsigned)(size_t)p;
}

// Async global->LDS copy, 16 bytes per lane. Tracked by ASYNCcnt.
__device__ __forceinline__ void async_copy_b128(unsigned dst_lds, const u16* src) {
    asm volatile("global_load_async_to_lds_b128 %0, %1, off"
                 :: "v"(dst_lds), "v"(src)
                 : "memory");
}
__device__ __forceinline__ void wait_asynccnt0() {
    asm volatile("s_wait_asynccnt 0x0" ::: "memory");
}

// ---------------------------------------------------------------------------
// fp32 -> bf16 (RNE)
// ---------------------------------------------------------------------------
__global__ void cvt_f32_bf16_kernel(const float* __restrict__ src,
                                    u16* __restrict__ dst, int n) {
    int i = blockIdx.x * blockDim.x + threadIdx.x;
    if (i < n) {
        unsigned u = __float_as_uint(src[i]);
        dst[i] = (u16)((u + 0x7FFFu + ((u >> 16) & 1u)) >> 16);
    }
}

// ---------------------------------------------------------------------------
// Zero state double-buffers + barrier counter (re-run every launch for
// deterministic graph replay).
// ---------------------------------------------------------------------------
__global__ void init_state_kernel(float* hbuf, float* cbuf, unsigned* cnt) {
    int i = blockIdx.x * blockDim.x + threadIdx.x;
    if (i < 2 * H) { hbuf[i] = 0.f; cbuf[i] = 0.f; }
    if (i == 0) *cnt = 0u;
}

// ---------------------------------------------------------------------------
// WMMA GEMM: C[g] = X(1024x2048,bf16) @ W[g](2048x2048,bf16) + bias[g]
//
// 256 threads = 8 waves as 4(M) x 2(N); block tile 64x64; each wave owns one
// 16-row A fragment and two 16-col B fragments -> 2 WMMA per wave per K-step.
// LDS double buffer (2 x (A 64x32 + B^T 64x32) bf16 = 16KB):
//   - A tile is a straight copy: global_load_async_to_lds_b128 (no VGPR bounce)
//   - B tile needs a transpose: regs + ds_store_b16 into [n][k] layout
// One s_barrier per K-iteration; tile i+1 traffic overlaps tile i WMMAs.
// Fragment layout per ISA (16-bit): lanes 0-15 hold K 0..7 & 16..23,
// lanes 16-31 hold K 8..15 & 24..31 -> two b128 LDS reads per fragment.
// ---------------------------------------------------------------------------
__global__ void __launch_bounds__(256)
gemm_x_wmma_kernel(const u16* __restrict__ Xb, const u16* __restrict__ Wxall,
                   const float* __restrict__ bi, const float* __restrict__ bo,
                   const float* __restrict__ bc, float* __restrict__ XWall) {
    __shared__ __attribute__((aligned(16))) u16 bufA[2][64][32];  // [m][k]
    __shared__ __attribute__((aligned(16))) u16 bufB[2][64][32];  // [n][k]

    const int gate = blockIdx.z;
    const u16* W = Wxall + (size_t)gate * H * H;
    const float* bias = (gate == 0) ? bi : ((gate == 1) ? bo : bc);
    float* C = XWall + (size_t)gate * S * H;

    const int tid  = threadIdx.x;
    const int wid  = tid >> 5;           // wave 0..7
    const int lane = tid & 31;
    const int wm   = wid >> 1;           // 0..3  -> M offset wm*16
    const int wn   = wid & 1;            // 0..1  -> N offset wn*32
    const int mtile = blockIdx.y * 64;
    const int ntile = blockIdx.x * 64;

    const int mrow  = (wm << 4) + (lane & 15);
    const int nrow0 = (wn << 5) + (lane & 15);
    const int nrow1 = nrow0 + 16;
    const int fkk   = (lane >> 4) << 3;  // 0 or 8

    // cooperative staging coords
    const int ar = tid >> 2, ak = (tid & 3) << 3;   // A: 64 rows x 4 b128 chunks
    const int bk = tid >> 3, bn = (tid & 7) << 3;   // B: 32 k-rows x 8 b128 chunks

    // ---- prologue: stage tile 0 into buffer 0
    async_copy_b128(lds_off(&bufA[0][ar][ak]),
                    Xb + (size_t)(mtile + ar) * H + ak);
    U128 breg = *(const U128*)(W + (size_t)bk * H + ntile + bn);
    {
        const u16* bh = (const u16*)&breg;
#pragma unroll
        for (int e = 0; e < 8; ++e) bufB[0][bn + e][bk] = bh[e];
    }
    wait_asynccnt0();
    __syncthreads();

    v8f acc0 = {}, acc1 = {};
    constexpr int NITER = H / 32;

    for (int i = 0; i < NITER; ++i) {
        const int cur = i & 1, nxt = cur ^ 1;
        const bool more = (i + 1) < NITER;
        const int kn = (i + 1) << 5;

        if (more) {
            async_copy_b128(lds_off(&bufA[nxt][ar][ak]),
                            Xb + (size_t)(mtile + ar) * H + kn + ak);
            breg = *(const U128*)(W + (size_t)(kn + bk) * H + ntile + bn);
        }

        FragU fa, f0, f1;
        fa.u[0] = *(const U128*)&bufA[cur][mrow][fkk];
        fa.u[1] = *(const U128*)&bufA[cur][mrow][fkk + 16];
        f0.u[0] = *(const U128*)&bufB[cur][nrow0][fkk];
        f0.u[1] = *(const U128*)&bufB[cur][nrow0][fkk + 16];
        f1.u[0] = *(const U128*)&bufB[cur][nrow1][fkk];
        f1.u[1] = *(const U128*)&bufB[cur][nrow1][fkk + 16];
        acc0 = __builtin_amdgcn_wmma_f32_16x16x32_bf16(
            false, fa.v, false, f0.v, (short)0, acc0, false, false);
        acc1 = __builtin_amdgcn_wmma_f32_16x16x32_bf16(
            false, fa.v, false, f1.v, (short)0, acc1, false, false);

        if (more) {
            const u16* bh = (const u16*)&breg;
#pragma unroll
            for (int e = 0; e < 8; ++e) bufB[nxt][bn + e][bk] = bh[e];
            wait_asynccnt0();
        }
        __syncthreads();
    }

    // C/D layout: VGPR r -> M = r + 8*(lane>=16), N = lane&15
    const int n0 = ntile + (wn << 5) + (lane & 15);
    const int mb = mtile + (wm << 4) + ((lane >> 4) << 3);
    const float bv0 = bias[n0];
    const float bv1 = bias[n0 + 16];
#pragma unroll
    for (int r = 0; r < 8; ++r) {
        C[(size_t)(mb + r) * H + n0]      = acc0[r] + bv0;
        C[(size_t)(mb + r) * H + n0 + 16] = acc1[r] + bv1;
    }
}

// ---------------------------------------------------------------------------
// Grid-wide barrier: monotonic counter, release add / acquire spin.
// ---------------------------------------------------------------------------
__device__ __forceinline__ void grid_sync(unsigned* cnt, unsigned& goal) {
    __threadfence();
    __syncthreads();
    if (threadIdx.x == 0) {
        goal += gridDim.x;
        __hip_atomic_fetch_add(cnt, 1u, __ATOMIC_RELEASE, __HIP_MEMORY_SCOPE_AGENT);
        while (__hip_atomic_load(cnt, __ATOMIC_ACQUIRE, __HIP_MEMORY_SCOPE_AGENT) < goal)
            __builtin_amdgcn_s_sleep(1);
        __threadfence();
    }
    __syncthreads();
}

// ---------------------------------------------------------------------------
// Persistent recurrent kernel. NB blocks x 256 threads; block b owns columns
// [32b, 32b+32). Per step: phase 1 forms i_t, w_t, c_t (3 matvec slices),
// grid barrier, phase 2 forms o_t, h_t (2 matvec slices), grid barrier.
// State vectors double-buffered in global memory; staged per-step into LDS.
// bf16 weights keep the whole recurrent weight set L2-resident (5x8MB << 192MB)
// and halve per-step L2 traffic vs fp32.
// ---------------------------------------------------------------------------
__global__ void __launch_bounds__(256)
lstm_recurrent_kernel(const float* __restrict__ XWi, const float* __restrict__ XWo,
                      const float* __restrict__ XWc,
                      const u16* __restrict__ Wh2i, const u16* __restrict__ Wc2i,
                      const u16* __restrict__ Wh2o, const u16* __restrict__ Wc2o,
                      const u16* __restrict__ Wh2c,
                      float* __restrict__ hbuf, float* __restrict__ cbuf,
                      unsigned* cnt,
                      float* __restrict__ outH, float* __restrict__ outC) {
    __shared__ float sh_h[H];
    __shared__ float sh_c[H];       // c_{t-1} in phase 1, c_t in phase 2
    __shared__ float red_a[8][32];
    __shared__ float red_b[8][32];

    const int tid  = threadIdx.x;
    const int part = tid >> 5;               // k-partition 0..7
    const int cl   = tid & 31;               // column lane
    const int col  = blockIdx.x * 32 + cl;
    const int kbase = part * (H / 8);
    unsigned goal = 0;

    for (int t = 0; t < S; ++t) {
        const int r = t & 1, w = r ^ 1;
        const float* hr = hbuf + (size_t)r * H;
        const float* cr = cbuf + (size_t)r * H;
        float* cw = cbuf + (size_t)w * H;
        float* hw = hbuf + (size_t)w * H;

        for (int i = tid; i < H; i += 256) { sh_h[i] = hr[i]; sh_c[i] = cr[i]; }
        __syncthreads();

        // ---- phase 1: i_t = sigma(xWi + h h2i + c c2i); w_t = tanh(xWc + h h2c)
        float ai = 0.f, ac = 0.f;
        for (int kk = 0; kk < H / 8; ++kk) {
            const int k = kbase + kk;
            const float hk = sh_h[k], ck = sh_c[k];   // LDS broadcast
            const size_t wi = (size_t)k * H + col;     // 64B segment per group
            ai = __builtin_fmaf(hk, bf2f(Wh2i[wi]), ai);
            ai = __builtin_fmaf(ck, bf2f(Wc2i[wi]), ai);
            ac = __builtin_fmaf(hk, bf2f(Wh2c[wi]), ac);
        }
        red_a[part][cl] = ai;
        red_b[part][cl] = ac;
        __syncthreads();
        if (tid < 32) {
            float si = XWi[(size_t)t * H + col];
            float sc = XWc[(size_t)t * H + col];
#pragma unroll
            for (int p = 0; p < 8; ++p) { si += red_a[p][cl]; sc += red_b[p][cl]; }
            const float ii = 1.f / (1.f + __expf(-si));
            const float wt = tanhf(sc);
            const float ct = ii * wt + (1.f - ii) * sh_c[col];
            cw[col] = ct;
            outC[(size_t)t * H + col] = ct;
        }
        grid_sync(cnt, goal);

        // ---- phase 2: o_t = sigma(xWo + c_t c2o + h h2o); h_t = o_t tanh(c_t)
        for (int i = tid; i < H; i += 256) sh_c[i] = cw[i];
        __syncthreads();
        float ao = 0.f;
        for (int kk = 0; kk < H / 8; ++kk) {
            const int k = kbase + kk;
            const size_t wi = (size_t)k * H + col;
            ao = __builtin_fmaf(sh_h[k], bf2f(Wh2o[wi]), ao);
            ao = __builtin_fmaf(sh_c[k], bf2f(Wc2o[wi]), ao);
        }
        red_a[part][cl] = ao;
        __syncthreads();
        if (tid < 32) {
            float so = XWo[(size_t)t * H + col];
#pragma unroll
            for (int p = 0; p < 8; ++p) so += red_a[p][cl];
            const float oo = 1.f / (1.f + __expf(-so));
            const float ht = oo * tanhf(sh_c[col]);
            hw[col] = ht;
            outH[(size_t)t * H + col] = ht;
        }
        grid_sync(cnt, goal);
    }
}

// ---------------------------------------------------------------------------
// Launch
// ---------------------------------------------------------------------------
extern "C" void kernel_launch(void* const* d_in, const int* in_sizes, int n_in,
                              void* d_out, int out_size, void* d_ws, size_t ws_size,
                              hipStream_t stream) {
    const float* X   = (const float*)d_in[0];
    const float* x2i = (const float*)d_in[1];
    const float* h2i = (const float*)d_in[2];
    const float* c2i = (const float*)d_in[3];
    const float* bi  = (const float*)d_in[4];
    const float* x2o = (const float*)d_in[5];
    const float* h2o = (const float*)d_in[6];
    const float* c2o = (const float*)d_in[7];
    const float* bo  = (const float*)d_in[8];
    const float* x2c = (const float*)d_in[9];
    const float* h2c = (const float*)d_in[10];
    const float* bc  = (const float*)d_in[11];

    char* ws = (char*)d_ws;
    const size_t Mi = (size_t)1 << 20;
    const size_t NW = (size_t)H * H;           // 4M elements per weight matrix
    u16* Xb   = (u16*)(ws + 0);                // 4  MiB
    u16* Wx   = (u16*)(ws + 4 * Mi);           // 24 MiB (x2i, x2o, x2c)
    u16* Wh2i = (u16*)(ws + 28 * Mi);          // 8 MiB each below
    u16* Wc2i = (u16*)(ws + 36 * Mi);
    u16* Wh2o = (u16*)(ws + 44 * Mi);
    u16* Wc2o = (u16*)(ws + 52 * Mi);
    u16* Wh2c = (u16*)(ws + 60 * Mi);
    float* XW = (float*)(ws + 68 * Mi);        // 24 MiB (XWi, XWo, XWc)
    float* hbuf = (float*)(ws + 92 * Mi);      // 2 x H double buffer
    float* cbuf = (float*)(ws + 92 * Mi + 16384);
    unsigned* cnt = (unsigned*)(ws + 92 * Mi + 32768);

    const int nX = S * H;
    cvt_f32_bf16_kernel<<<(nX + 255) / 256, 256, 0, stream>>>(X, Xb, nX);
    const int nW = (int)NW;
    cvt_f32_bf16_kernel<<<(nW + 255) / 256, 256, 0, stream>>>(x2i, Wx + 0 * NW, nW);
    cvt_f32_bf16_kernel<<<(nW + 255) / 256, 256, 0, stream>>>(x2o, Wx + 1 * NW, nW);
    cvt_f32_bf16_kernel<<<(nW + 255) / 256, 256, 0, stream>>>(x2c, Wx + 2 * NW, nW);
    cvt_f32_bf16_kernel<<<(nW + 255) / 256, 256, 0, stream>>>(h2i, Wh2i, nW);
    cvt_f32_bf16_kernel<<<(nW + 255) / 256, 256, 0, stream>>>(c2i, Wc2i, nW);
    cvt_f32_bf16_kernel<<<(nW + 255) / 256, 256, 0, stream>>>(h2o, Wh2o, nW);
    cvt_f32_bf16_kernel<<<(nW + 255) / 256, 256, 0, stream>>>(c2o, Wc2o, nW);
    cvt_f32_bf16_kernel<<<(nW + 255) / 256, 256, 0, stream>>>(h2c, Wh2c, nW);

    dim3 gg(H / 64, S / 64, 3);
    gemm_x_wmma_kernel<<<gg, 256, 0, stream>>>(Xb, Wx, bi, bo, bc, XW);

    init_state_kernel<<<(2 * H + 255) / 256, 256, 0, stream>>>(hbuf, cbuf, cnt);

    float* outH = (float*)d_out;
    float* outC = outH + (size_t)S * H;
    lstm_recurrent_kernel<<<NB, 256, 0, stream>>>(
        XW, XW + (size_t)S * H, XW + 2 * (size_t)S * H,
        Wh2i, Wc2i, Wh2o, Wc2o, Wh2c, hbuf, cbuf, cnt, outH, outC);
}
